// Network_74259984548393
// MI455X (gfx1250) — compile-verified, matrix operand order checked by reading, not run on hardware
//
#include <hip/hip_runtime.h>
#include <hip/hip_bf16.h>
#include <math.h>

// ---------------------------------------------------------------------------
// XNOR-Net MLP forward for MI455X (gfx1250, wave32, WMMA).
//   h = x @ (sign(W)*alpha).T + b  ==  (x @ sign(W).T) * alpha[col] + b[col]
// GEMMs run on v_wmma_f32_16x16x32_bf16 with fp32 accumulation; sign weights
// are exact in bf16 (+1/-1/0); activations are bf16.
// ---------------------------------------------------------------------------

typedef __attribute__((ext_vector_type(16))) __bf16 v16bf;
typedef __attribute__((ext_vector_type(8)))  float  v8f;

#define BATCH   16384
#define KPAD    1024
#define EPSBN   1e-5f

// ---------------------------------------------------------------------------
// Convert x (16384 x 784 fp32) -> bf16 padded to 16384 x 1024 (zeros in pad).
// ---------------------------------------------------------------------------
__global__ __launch_bounds__(256) void cvt_x_kernel(const float* __restrict__ x,
                                                    __bf16* __restrict__ A) {
    int idx = blockIdx.x * 256 + threadIdx.x;       // over 16384*1024
    int row = idx >> 10;
    int col = idx & 1023;
    float v = (col < 784) ? x[row * 784 + col] : 0.0f;
    A[idx] = (__bf16)v;
}

// ---------------------------------------------------------------------------
// Per-output-row binarization: alpha[o] = mean(|W[o,:]|) over Kreal,
// Wb[o, 0..1023] = sign(W[o,c]) as bf16 (+1/-1/0), zero past Kreal / Orows.
// One block per (padded) output row.
// ---------------------------------------------------------------------------
__global__ __launch_bounds__(256) void binarize_kernel(const float* __restrict__ W,
                                                       __bf16* __restrict__ Wb,
                                                       float* __restrict__ alpha,
                                                       int Kreal, int Orows) {
    int o = blockIdx.x;
    __shared__ float red[256];
    float s = 0.0f;
    if (o < Orows) {
        for (int c = threadIdx.x; c < Kreal; c += 256)
            s += fabsf(W[(long)o * Kreal + c]);
    }
    red[threadIdx.x] = s;
    __syncthreads();
    for (int off = 128; off > 0; off >>= 1) {
        if (threadIdx.x < off) red[threadIdx.x] += red[threadIdx.x + off];
        __syncthreads();
    }
    if (threadIdx.x == 0)
        alpha[o] = (o < Orows) ? red[0] / (float)Kreal : 0.0f;

    for (int c = threadIdx.x; c < KPAD; c += 256) {
        float w = (o < Orows && c < Kreal) ? W[(long)o * Kreal + c] : 0.0f;
        float sgn = (w > 0.0f) ? 1.0f : ((w < 0.0f) ? -1.0f : 0.0f);
        Wb[(long)o * KPAD + c] = (__bf16)sgn;
    }
}

// ---------------------------------------------------------------------------
// Main GEMM: H[16384 x 1024] = A[16384 x 1024](bf16) @ Wb^T * alpha + bias.
// Block tile 128x128, 8 waves, each wave 32(M) x 64(N) = 2x4 WMMA tiles.
// A fragment layout used here (16-bit A 16x32, not the ISA table order): lane = (K/16)*16 + row,
// each lane holds 16 contiguous bf16 (32 B).  W is (N x K) row-major, and the
// B (KxN) fragment layout mirrors A with N in place of M, so W rows load
// identically to A rows.
// ---------------------------------------------------------------------------
__global__ __launch_bounds__(256) void gemm_bin128_kernel(
        const __bf16* __restrict__ A, const __bf16* __restrict__ Wb,
        const float* __restrict__ alpha, const float* __restrict__ bias,
        float* __restrict__ H) {
    const int K = KPAD;
    int w     = threadIdx.x >> 5;
    int lane  = threadIdx.x & 31;
    int wr    = w >> 1;          // 0..3  -> M sub-tile
    int wc    = w & 1;           // 0..1  -> N sub-tile
    int lrow  = lane & 15;
    int lhalf = lane >> 4;

    long baseM = (long)blockIdx.x * 128 + wr * 32;
    long baseN = (long)blockIdx.y * 128 + wc * 64;

    v8f acc[2][4];
#pragma unroll
    for (int i = 0; i < 2; ++i)
#pragma unroll
        for (int j = 0; j < 4; ++j) acc[i][j] = v8f{};

    const __bf16* pa = A  + (baseM + lrow) * K + lhalf * 16;
    const __bf16* pb = Wb + (baseN + lrow) * K + lhalf * 16;

    for (int k = 0; k < K; k += 32) {
        // hint-prefetch next K-slab of A (emits global_prefetch_b8)
        __builtin_prefetch(pa + k + 32, 0, 1);

        v16bf af[2], bfv[4];
#pragma unroll
        for (int i = 0; i < 2; ++i)
            af[i] = *(const v16bf*)(pa + (long)i * 16 * K + k);
#pragma unroll
        for (int j = 0; j < 4; ++j)
            bfv[j] = *(const v16bf*)(pb + (long)j * 16 * K + k);

#pragma unroll
        for (int i = 0; i < 2; ++i)
#pragma unroll
            for (int j = 0; j < 4; ++j)
                acc[i][j] = __builtin_amdgcn_wmma_f32_16x16x32_bf16(
                    false, af[i], false, bfv[j], (short)0, acc[i][j], false, false);
    }

    // Epilogue: C layout — VGPR r: lanes 0-15 -> M=r (N=lane), lanes 16-31 -> M=r+8.
#pragma unroll
    for (int j = 0; j < 4; ++j) {
        long col = baseN + j * 16 + lrow;
        float al = alpha[col];
        float bi = bias[col];
#pragma unroll
        for (int i = 0; i < 2; ++i) {
#pragma unroll
            for (int r = 0; r < 8; ++r) {
                long row = baseM + i * 16 + lhalf * 8 + r;
                H[row * KPAD + col] = acc[i][j][r] * al + bi;
            }
        }
    }
}

// ---------------------------------------------------------------------------
// Final layer GEMM: H5[16384 x 16] (cols 10..15 are padded zeros: alpha=0).
// Each wave does one 16x16 tile; block of 8 waves covers 128 rows.
// ---------------------------------------------------------------------------
__global__ __launch_bounds__(256) void gemm_bin_n16_kernel(
        const __bf16* __restrict__ A, const __bf16* __restrict__ Wb,
        const float* __restrict__ alpha, const float* __restrict__ bias,
        float* __restrict__ H5) {
    const int K = KPAD;
    int w     = threadIdx.x >> 5;
    int lane  = threadIdx.x & 31;
    int lrow  = lane & 15;
    int lhalf = lane >> 4;

    long baseM = (long)blockIdx.x * 128 + w * 16;
    const __bf16* pa = A  + (baseM + lrow) * K + lhalf * 16;
    const __bf16* pb = Wb + (long)lrow * K + lhalf * 16;

    v8f acc = v8f{};
    for (int k = 0; k < K; k += 32) {
        v16bf av = *(const v16bf*)(pa + k);
        v16bf bv = *(const v16bf*)(pb + k);
        acc = __builtin_amdgcn_wmma_f32_16x16x32_bf16(
            false, av, false, bv, (short)0, acc, false, false);
    }

    int col = lrow;
    float al = alpha[col];
    float bi = (col < 10) ? bias[col] : 0.0f;
#pragma unroll
    for (int r = 0; r < 8; ++r) {
        long row = baseM + lhalf * 8 + r;
        H5[row * 16 + col] = acc[r] * al + bi;
    }
}

// ---------------------------------------------------------------------------
// Column statistics for training-mode BatchNorm: sum and sum-of-squares per
// column.  Grid = (4 col-blocks) x (64 row-blocks); register accumulation over
// 256 rows then two float atomics per thread.  stats[0..1023]=sum,
// stats[1024..2047]=sumsq (zeroed via hipMemsetAsync before launch).
// ---------------------------------------------------------------------------
__global__ __launch_bounds__(256) void colstats_kernel(const float* __restrict__ H,
                                                       float* __restrict__ stats) {
    int cb  = blockIdx.x & 3;
    int rb  = blockIdx.x >> 2;
    int col = cb * 256 + threadIdx.x;
    float s = 0.0f, s2 = 0.0f;
    int r0 = rb * 256;
    for (int r = r0; r < r0 + 256; ++r) {
        float v = H[(long)r * KPAD + col];
        s  += v;
        s2 += v * v;
    }
    atomicAdd(&stats[col], s);
    atomicAdd(&stats[KPAD + col], s2);
}

// ---------------------------------------------------------------------------
// BN (batch stats, biased var) + ReLU + convert to bf16 activation.
// ---------------------------------------------------------------------------
__global__ __launch_bounds__(256) void bnrelu_kernel(const float* __restrict__ H,
                                                     const float* __restrict__ stats,
                                                     const float* __restrict__ g,
                                                     const float* __restrict__ be,
                                                     __bf16* __restrict__ out) {
    int idx = blockIdx.x * 256 + threadIdx.x;       // over 16384*1024
    int col = idx & 1023;
    const float invB = 1.0f / (float)BATCH;
    float mu  = stats[col] * invB;
    float var = stats[KPAD + col] * invB - mu * mu;
    float sc  = g[col] * rsqrtf(var + EPSBN);
    float v   = sc * (H[idx] - mu) + be[col];
    out[idx]  = (__bf16)fmaxf(v, 0.0f);
}

// ---------------------------------------------------------------------------
// log_softmax over the 10 valid columns of H5 (stride 16), one row per thread.
// ---------------------------------------------------------------------------
__global__ __launch_bounds__(256) void logsoftmax_kernel(const float* __restrict__ H5,
                                                         float* __restrict__ out) {
    int row = blockIdx.x * 256 + threadIdx.x;
    if (row >= BATCH) return;
    float v[10];
    float m = -INFINITY;
#pragma unroll
    for (int j = 0; j < 10; ++j) {
        v[j] = H5[row * 16 + j];
        m = fmaxf(m, v[j]);
    }
    float s = 0.0f;
#pragma unroll
    for (int j = 0; j < 10; ++j) s += __expf(v[j] - m);
    float lse = m + __logf(s);
#pragma unroll
    for (int j = 0; j < 10; ++j) out[row * 10 + j] = v[j] - lse;
}

// ---------------------------------------------------------------------------
// Host-side orchestration.  Inputs (all fp32):
//  0:x 1:W1 2:b1 3:W2 4:b2 5:W3 6:b3 7:W4 8:b4 9:W5 10:b5
//  11:g1 12:be1 13:g2 14:be2 15:g3 16:be3 17:g4 18:be4
// ---------------------------------------------------------------------------
extern "C" void kernel_launch(void* const* d_in, const int* in_sizes, int n_in,
                              void* d_out, int out_size, void* d_ws, size_t ws_size,
                              hipStream_t stream) {
    (void)in_sizes; (void)n_in; (void)out_size; (void)ws_size;

    const float* x = (const float*)d_in[0];
    const float* W[5]  = {(const float*)d_in[1], (const float*)d_in[3],
                          (const float*)d_in[5], (const float*)d_in[7],
                          (const float*)d_in[9]};
    const float* b[5]  = {(const float*)d_in[2], (const float*)d_in[4],
                          (const float*)d_in[6], (const float*)d_in[8],
                          (const float*)d_in[10]};
    const float* g[4]  = {(const float*)d_in[11], (const float*)d_in[13],
                          (const float*)d_in[15], (const float*)d_in[17]};
    const float* be[4] = {(const float*)d_in[12], (const float*)d_in[14],
                          (const float*)d_in[16], (const float*)d_in[18]};

    // Workspace layout (bytes):
    //   actA bf16 16384x1024  :  33,554,432
    //   actB bf16 16384x1024  :  33,554,432
    //   h    f32  16384x1024  :  67,108,864   (also reused as 16384x16 h5)
    //   Wb   bf16 1024x1024   :   2,097,152
    //   alpha f32 1024        :       4,096
    //   stats f32 2048        :       8,192
    char* ws = (char*)d_ws;
    __bf16* actA  = (__bf16*)(ws);
    __bf16* actB  = (__bf16*)(ws + 33554432);
    float*  h     = (float*)(ws + 67108864);
    __bf16* Wb    = (__bf16*)(ws + 134217728);
    float*  alpha = (float*)(ws + 136314880);
    float*  stats = (float*)(ws + 136318976);

    // x -> bf16 padded activation
    cvt_x_kernel<<<(BATCH * KPAD) / 256, 256, 0, stream>>>(x, actA);

    __bf16* src = actA;
    __bf16* dst = actB;
    for (int i = 0; i < 4; ++i) {
        int Kreal = (i == 0) ? 784 : 1024;
        binarize_kernel<<<1024, 256, 0, stream>>>(W[i], Wb, alpha, Kreal, 1024);
        dim3 grid(BATCH / 128, KPAD / 128);
        gemm_bin128_kernel<<<grid, 256, 0, stream>>>(src, Wb, alpha, b[i], h);
        hipMemsetAsync(stats, 0, 2048 * sizeof(float), stream);
        colstats_kernel<<<256, 256, 0, stream>>>(h, stats);
        bnrelu_kernel<<<(BATCH * KPAD) / 256, 256, 0, stream>>>(h, stats, g[i], be[i], dst);
        __bf16* t = src; src = dst; dst = t;
    }

    // Layer 5: N=10 padded to 16 (alpha=0 in pad rows), then log_softmax.
    binarize_kernel<<<16, 256, 0, stream>>>(W[4], Wb, alpha, 1024, 10);
    gemm_bin_n16_kernel<<<BATCH / 128, 256, 0, stream>>>(src, Wb, alpha, b[4], h);
    logsoftmax_kernel<<<BATCH / 256, 256, 0, stream>>>(h, (float*)d_out);
}
